// vertex2hyperedge_46059229282783
// MI455X (gfx1250) — matrix operand order repaired and checked by reading, not compiled, not run on hardware
//
#include <hip/hip_runtime.h>

// Problem constants (match reference)
#define B_    8
#define E_    2048
#define N_    2048
#define F_    128
#define ETILE 32          // E-rows per workgroup
#define KSTEP 32          // K per WMMA step (bf16 16x16x32)

typedef __attribute__((ext_vector_type(16))) __bf16 v16bf;
typedef __attribute__((ext_vector_type(8)))  __bf16 v8bf;
typedef __attribute__((ext_vector_type(8)))  float  v8f;
typedef __attribute__((ext_vector_type(4)))  unsigned int u32x4;
typedef __attribute__((ext_vector_type(8)))  int i32x8;
typedef __attribute__((ext_vector_type(4)))  int i32x4;

// Build a 16-bit A-fragment (16x32) per the CDNA5 ISA layout:
// lanes 0-15 hold K{0..7,16..23}, lanes 16-31 hold K{8..15,24..31} of row (lane&15).
static __device__ __forceinline__ v16bf make_afrag(const __bf16* p0, const __bf16* p1) {
  v8bf lo = *(const v8bf*)p0;
  v8bf hi = *(const v8bf*)p1;
  v16bf r;
#pragma unroll
  for (int i = 0; i < 8; ++i) { r[i] = lo[i]; r[i + 8] = hi[i]; }
  return r;
}

// --- Tensor Data Mover: load a 2D tile [F_ rows x KSTEP elems] of bf16 from the
// transposed tensor xbf[b] (dim0 = N_ inner, dim1 = F_ rows, stride0 = N_) into
// LDS at lds_off. TDM writes tile rows back-to-back => exactly xT[f][k] with
// row stride KSTEP*2 = 64 B. Tracked with TENSORcnt.
static __device__ __forceinline__ void tdm_load_xslab(const __bf16* gsrc, unsigned lds_off) {
  unsigned long long ga = (unsigned long long)(uintptr_t)gsrc;
  u32x4 g0;
  g0[0] = 1u;                                                  // count=1, user mode, no gather
  g0[1] = lds_off;                                             // D#.lds_addr (bytes)
  g0[2] = (unsigned)ga;                                        // global_addr[31:0]
  g0[3] = (unsigned)((ga >> 32) & 0x01ffffffu) | (2u << 30);   // global_addr[56:32] | type=2
  i32x8 g1;
  g1[0] = 1 << 16;                                             // data_size=1 (2 bytes)
  g1[1] = (N_ & 0xffff) << 16;                                 // tensor_dim0[15:0]
  g1[2] = (N_ >> 16) | ((F_ & 0xffff) << 16);                  // tensor_dim0 hi | tensor_dim1 lo
  g1[3] = (F_ >> 16) | (KSTEP << 16);                          // tensor_dim1 hi | tile_dim0
  g1[4] = F_;                                                  // tile_dim1=F_, tile_dim2=0
  g1[5] = N_;                                                  // tensor_dim0_stride[31:0]
  g1[6] = 0;                                                   // stride0 hi | stride1 lo
  g1[7] = 0;                                                   // stride1 hi
  i32x4 gz = {0, 0, 0, 0};
#if defined(__clang_major__) && (__clang_major__ >= 23)
  i32x8 gz8 = {0, 0, 0, 0, 0, 0, 0, 0};
  __builtin_amdgcn_tensor_load_to_lds(g0, g1, gz, gz, gz8, 0);
#else
  __builtin_amdgcn_tensor_load_to_lds(g0, g1, gz, gz, 0);
#endif
}

// --- Pre-pass: xbf[b][f][n] = (bf16)x[b][n][f], LDS-tiled transpose so both
// global read and write are coalesced. Tiny: 8 MiB read + 4 MiB write.
__global__ __launch_bounds__(256)
void v2h_cvt_x_kernel(const float* __restrict__ x, __bf16* __restrict__ xbf) {
  __shared__ float tile[32][65];            // [f][n], padded vs bank conflicts
  const int blk = blockIdx.x;               // B_ * (N_/64) * (F_/32) = 1024
  const int fb  = blk & 3;
  const int nb  = (blk >> 2) & 31;
  const int b   = blk >> 7;
  const int f0 = fb * 32, n0 = nb * 64;
  const float* xb = x + ((size_t)b * N_ + n0) * F_ + f0;
#pragma unroll
  for (int i = 0; i < 8; ++i) {
    int idx = threadIdx.x + 256 * i;        // 0..2047
    int n = idx >> 5, f = idx & 31;         // consecutive lanes -> consecutive f (coalesced read)
    tile[f][n] = xb[(size_t)n * F_ + f];
  }
  __syncthreads();
  __bf16* ob = xbf + ((size_t)b * F_ + f0) * N_ + n0;
#pragma unroll
  for (int i = 0; i < 8; ++i) {
    int idx = threadIdx.x + 256 * i;
    int f = idx >> 6, n = idx & 63;         // consecutive lanes -> consecutive n (coalesced write)
    ob[(size_t)f * N_ + n] = (__bf16)tile[f][n];
  }
}

template <bool USE_TDM>
__global__ __launch_bounds__(256, 2)
void v2h_wmma_kernel(const float* __restrict__ x,     // [B,N,F]
                     const float* __restrict__ adj,   // [B,E,N]
                     const float* __restrict__ W,     // [F,F]
                     const float* __restrict__ bias,  // [F]
                     const __bf16* __restrict__ xbf,  // [B,F,N] bf16 (TDM path)
                     float* __restrict__ out) {       // [B,E,F]
  __shared__ __align__(32) __bf16 maskA[ETILE * KSTEP];   // [e][k] 0/1 mask (bf16-exact)
  __shared__ __align__(64) __bf16 xT[2][F_ * KSTEP];      // [f][k], double-buffered for TDM
  __shared__ __align__(32) __bf16 wB[F_ * F_];            // [f_out][f_in]
  __shared__ __align__(32) __bf16 sA[ETILE * F_];         // message_vertex tile (S/norm)
  __shared__ float normv[ETILE];

  const int tid  = threadIdx.x;
  const int wave = tid >> 5;
  const int lane = tid & 31;
  const int hi   = lane >> 4;
  const int l15  = lane & 15;
  // scalar (SGPR) wave id -> guarantees a scalar branch around TDM issue
  const int s_wave = __builtin_amdgcn_readfirstlane(tid) >> 5;

  const int tile = blockIdx.x;
  const int b    = tile / (E_ / ETILE);
  const int e0   = (tile % (E_ / ETILE)) * ETILE;

  const float* adjBase = adj + ((size_t)b * E_ + e0) * N_;
  const float* xBase   = x   + (size_t)b * N_ * F_;
  const __bf16* xbfBase = xbf + (size_t)b * F_ * N_;

  const unsigned lds_x0 = (unsigned)(uintptr_t)(&xT[0][0]);
  const unsigned lds_x1 = (unsigned)(uintptr_t)(&xT[1][0]);

  // Kick off the first TDM slab before doing anything else (overlaps W preload).
  if (USE_TDM) {
    if (s_wave == 0) tdm_load_xslab(xbfBase, lds_x0);
  }

  // Preload W into LDS as bf16 (L2-resident, shared by all WGs)
  for (int i = tid; i < F_ * F_; i += 256) wB[i] = (__bf16)W[i];
  if (tid < ETILE) normv[tid] = 0.0f;

  // adj staging map: each thread owns 4 consecutive k of one fixed e-row per step
  const int a_row  = tid >> 3;          // 0..31
  const int a_col0 = (tid & 7) * 4;     // 0..28

  v8f acc0 = {}; v8f acc1 = {};
  float cnt = 0.0f;

  __syncthreads();

  for (int k0 = 0; k0 < N_; k0 += KSTEP) {
    const int cur = USE_TDM ? ((k0 >> 5) & 1) : 0;

    // Issue the NEXT slab's DMA first so it overlaps this step's work.
    if (USE_TDM) {
      if (s_wave == 0 && (k0 + KSTEP) < N_)
        tdm_load_xslab(xbfBase + (k0 + KSTEP), cur ? lds_x0 : lds_x1);
    }

    // --- stage mask tile: float4 load, compare, packed bf16 store (1x b64) ---
    float4 av = *(const float4*)(adjBase + (size_t)a_row * N_ + k0 + a_col0);
    float m0 = (av.x == -1.0f) ? 1.0f : 0.0f;
    float m1 = (av.y == -1.0f) ? 1.0f : 0.0f;
    float m2 = (av.z == -1.0f) ? 1.0f : 0.0f;
    float m3 = (av.w == -1.0f) ? 1.0f : 0.0f;
    union { __bf16 h[4]; unsigned long long u; } pk;
    pk.h[0] = (__bf16)m0; pk.h[1] = (__bf16)m1;
    pk.h[2] = (__bf16)m2; pk.h[3] = (__bf16)m3;
    *(unsigned long long*)&maskA[a_row * KSTEP + a_col0] = pk.u;
    cnt += m0 + m1 + m2 + m3;

    if (!USE_TDM) {
      // fallback staging: float4 reads of x[k][f], transposed bf16 stores
#pragma unroll
      for (int i = 0; i < 4; ++i) {
        int v  = tid + 256 * i;           // 0..1023 float4s
        int kk = v >> 5;                  // 0..31
        int f0 = (v & 31) * 4;
        float4 xv = *(const float4*)(xBase + (size_t)(k0 + kk) * F_ + f0);
        xT[0][(f0 + 0) * KSTEP + kk] = (__bf16)xv.x;
        xT[0][(f0 + 1) * KSTEP + kk] = (__bf16)xv.y;
        xT[0][(f0 + 2) * KSTEP + kk] = (__bf16)xv.z;
        xT[0][(f0 + 3) * KSTEP + kk] = (__bf16)xv.w;
      }
    }

    // prefetch next adj tile (gfx1250 global_prefetch_b8)
    if (k0 + KSTEP < N_)
      __builtin_prefetch(adjBase + (size_t)a_row * N_ + (k0 + KSTEP) + a_col0, 0, 1);

    if (USE_TDM) {
      // slab `cur` complete; keep the just-issued slab in flight
      if ((k0 + KSTEP) < N_) __builtin_amdgcn_s_wait_tensorcnt(1);
      else                   __builtin_amdgcn_s_wait_tensorcnt(0);
    }
    __syncthreads();

    // --- fragments + 2x WMMA (E-rows 0-15 and 16-31 of the tile) ---
    const __bf16* xcur = &xT[cur][0];
    v16bf bf = make_afrag(&xcur[(16 * wave + l15) * KSTEP + hi * 8],
                          &xcur[(16 * wave + l15) * KSTEP + 16 + hi * 8]);
    v16bf a0 = make_afrag(&maskA[l15 * KSTEP + hi * 8],
                          &maskA[l15 * KSTEP + 16 + hi * 8]);
    v16bf a1 = make_afrag(&maskA[(16 + l15) * KSTEP + hi * 8],
                          &maskA[(16 + l15) * KSTEP + 16 + hi * 8]);
    acc0 = __builtin_amdgcn_wmma_f32_16x16x32_bf16(false, a0, false, bf, (short)0, acc0, false, false);
    acc1 = __builtin_amdgcn_wmma_f32_16x16x32_bf16(false, a1, false, bf, (short)0, acc1, false, false);

    __syncthreads();
  }

  // --- per-row mask counts (norm) ---
  atomicAdd(&normv[a_row], cnt);
  __syncthreads();

  // --- S/norm -> LDS bf16 (C/D layout: VGPR i is row i (lanes<16) or 8+i) ---
#pragma unroll
  for (int i = 0; i < 8; ++i) {
    int r0 = i + hi * 8;
    float n0 = normv[r0];      n0 = (n0 == 0.0f) ? 1.0f : n0;
    float n1 = normv[16 + r0]; n1 = (n1 == 0.0f) ? 1.0f : n1;
    sA[r0 * F_ + 16 * wave + l15]        = (__bf16)(acc0[i] / n0);
    sA[(16 + r0) * F_ + 16 * wave + l15] = (__bf16)(acc1[i] / n1);
  }
  __syncthreads();

  // --- second matmul: out = relu(S @ W^T + b); B[k][n] = W[n][k] contiguous in wB ---
  v8f o0 = {}; v8f o1 = {};
#pragma unroll
  for (int kk = 0; kk < F_; kk += KSTEP) {
    v16bf bw = make_afrag(&wB[(16 * wave + l15) * F_ + kk + hi * 8],
                          &wB[(16 * wave + l15) * F_ + kk + 16 + hi * 8]);
    v16bf a0 = make_afrag(&sA[l15 * F_ + kk + hi * 8],
                          &sA[l15 * F_ + kk + 16 + hi * 8]);
    v16bf a1 = make_afrag(&sA[(16 + l15) * F_ + kk + hi * 8],
                          &sA[(16 + l15) * F_ + kk + 16 + hi * 8]);
    o0 = __builtin_amdgcn_wmma_f32_16x16x32_bf16(false, a0, false, bw, (short)0, o0, false, false);
    o1 = __builtin_amdgcn_wmma_f32_16x16x32_bf16(false, a1, false, bw, (short)0, o1, false, false);
  }

  // --- epilogue: +bias, relu, store ---
  float bv = bias[16 * wave + l15];
  float* outBase = out + ((size_t)b * E_ + e0) * F_;
#pragma unroll
  for (int i = 0; i < 8; ++i) {
    int r0 = i + hi * 8;
    float v0 = o0[i] + bv; v0 = v0 > 0.0f ? v0 : 0.0f;
    float v1 = o1[i] + bv; v1 = v1 > 0.0f ? v1 : 0.0f;
    outBase[(size_t)r0 * F_ + 16 * wave + l15]        = v0;
    outBase[(size_t)(16 + r0) * F_ + 16 * wave + l15] = v1;
  }
}

extern "C" void kernel_launch(void* const* d_in, const int* in_sizes, int n_in,
                              void* d_out, int out_size, void* d_ws, size_t ws_size,
                              hipStream_t stream) {
  (void)in_sizes; (void)n_in; (void)out_size;
  const float* x    = (const float*)d_in[0];  // node_embeddings [B,N,F]
  const float* adj  = (const float*)d_in[1];  // adj [B,E,N]
  const float* W    = (const float*)d_in[2];  // W [F,F]
  const float* bias = (const float*)d_in[3];  // b [F]
  float* out        = (float*)d_out;          // [B,E,F]

  const size_t xbf_bytes = (size_t)B_ * F_ * N_ * sizeof(__bf16);  // 4 MiB
  dim3 grid((B_ * E_) / ETILE);   // 512 workgroups
  dim3 block(256);                // 8 wave32 per WG

  if (ws_size >= xbf_bytes) {
    __bf16* xbf = (__bf16*)d_ws;
    hipLaunchKernelGGL(v2h_cvt_x_kernel, dim3(B_ * (N_ / 64) * (F_ / 32)), block, 0, stream,
                       x, xbf);
    hipLaunchKernelGGL((v2h_wmma_kernel<true>), grid, block, 0, stream,
                       x, adj, W, bias, (const __bf16*)xbf, out);
  } else {
    hipLaunchKernelGGL((v2h_wmma_kernel<false>), grid, block, 0, stream,
                       x, adj, W, bias, (const __bf16*)nullptr, out);
  }
}
